// LinearAttention_47304769798231
// MI455X (gfx1250) — matrix operand order repaired and checked by reading, not compiled, not run on hardware
//
#include <hip/hip_runtime.h>
#include <math.h>

// Causal linear attention (elu+1 feature map), chunked parallel formulation.
// B=2, T=4096, H=8, E=D=64. Chunk=64 -> all GEMMs are 64x64x64 done as
// 16x16x32 f16 WMMA tiles (f32 accum). Workspace: B*H*NC*(64*64+64) f32
// = 16.25 MB (per-chunk K^T V states + ksum, exclusive-scanned in place).
// LDS tiles are staged so every WMMA fragment load is 2x ds_load_b128.

typedef __attribute__((ext_vector_type(16))) _Float16 v16h;
typedef __attribute__((ext_vector_type(8)))  _Float16 v8h;
typedef __attribute__((ext_vector_type(8)))  float    v8f;

#define Bdim  2
#define Tdim  4096
#define Hdim  8
#define Edim  64
#define CHUNK 64
#define NC    (Tdim / CHUNK)       // 64 chunks per head
#define NH    (Bdim * Hdim)        // 16 heads
#define LDH   72                   // padded f16 LDS row stride (144 B, 16B-aligned rows)
#define SSIZE (Edim * Edim + Edim) // 4160 f32 per chunk state
#define EPSC  1e-6f

__device__ __forceinline__ float phi_map(float x) {
    // elu(x)+1 == exp(x) for x<0, x+1 for x>=0
    return x > 0.f ? x + 1.f : __expf(x);
}

// ---- WMMA fragment loaders (wave32 layouts per CDNA5 ISA 7.12.2) ----------
// All loads are contiguous 16B-aligned v8h pairs -> ds_load_b128.

// A 16x32 f16 fragment from row-major matrix: element(m,k) = p[(m0+m)*ld + k0+k].
// Lane layout: a[0..7] = row[k0 + hi*8 .. +7], a[8..15] = row[k0+16+hi*8 .. +7].
__device__ __forceinline__ v16h load_a_row(const _Float16* p, int ld, int m0, int k0, int lane) {
    int hi = lane >> 4;
    const _Float16* row = p + (m0 + (lane & 15)) * ld + k0 + hi * 8;
    v8h lo = *(const v8h*)(row);
    v8h hh = *(const v8h*)(row + 16);
    return __builtin_shufflevector(lo, hh, 0, 1, 2, 3, 4, 5, 6, 7,
                                           8, 9, 10, 11, 12, 13, 14, 15);
}
// B 32x16 f16 fragment from the TRANSPOSE of an LDS matrix:
// element(k,n) = p[(n0+n)*ld + k0+k]  (lane reads k0+kh*8..+7 and k0+16+kh*8..+7).
__device__ __forceinline__ v16h load_b_col(const _Float16* p, int ld, int k0, int n0, int lane) {
    int kh = lane >> 4;
    const _Float16* row = p + (n0 + (lane & 15)) * ld + k0 + kh * 8;
    v8h lo = *(const v8h*)(row);
    v8h hh = *(const v8h*)(row + 16);
    return __builtin_shufflevector(lo, hh, 0, 1, 2, 3, 4, 5, 6, 7,
                                           8, 9, 10, 11, 12, 13, 14, 15);
}

#define WMMA_F16(acc, a, b) \
    __builtin_amdgcn_wmma_f32_16x16x32_f16(false, (a), false, (b), (short)0, (acc), false, false)

// ---- Pass 1: per-chunk S_c = phi(K)^T V  and  ksum_c ----------------------
__global__ __launch_bounds__(128) void la_chunk_sums(const float* __restrict__ Kg,
                                                     const float* __restrict__ Vg,
                                                     float* __restrict__ ws) {
    __shared__ alignas(16) _Float16 Kt[Edim * LDH];  // Kt[e][t] = phi(K)[t][e]
    __shared__ alignas(16) _Float16 Vt[Edim * LDH];  // Vt[d][t] = V[t][d]
    int blk = blockIdx.x;
    int c = blk % NC, bh = blk / NC;
    int b = bh / Hdim, h = bh % Hdim;
    int t0 = c * CHUNK;
    int tid = threadIdx.x;
    size_t base = ((size_t)(b * Tdim + t0) * Hdim + h) * Edim;

    // coalesced float4 global loads, transpose-scatter into LDS
    for (int i = tid; i < CHUNK * Edim / 4; i += 128) {
        int t = i >> 4, e4 = (i & 15) * 4;
        const float4 kq = *(const float4*)(Kg + base + (size_t)t * Hdim * Edim + e4);
        const float4 vq = *(const float4*)(Vg + base + (size_t)t * Hdim * Edim + e4);
        Kt[(e4 + 0) * LDH + t] = (_Float16)phi_map(kq.x);
        Kt[(e4 + 1) * LDH + t] = (_Float16)phi_map(kq.y);
        Kt[(e4 + 2) * LDH + t] = (_Float16)phi_map(kq.z);
        Kt[(e4 + 3) * LDH + t] = (_Float16)phi_map(kq.w);
        Vt[(e4 + 0) * LDH + t] = (_Float16)vq.x;
        Vt[(e4 + 1) * LDH + t] = (_Float16)vq.y;
        Vt[(e4 + 2) * LDH + t] = (_Float16)vq.z;
        Vt[(e4 + 3) * LDH + t] = (_Float16)vq.w;
    }
    __syncthreads();

    int wave = tid >> 5, lane = tid & 31;
    float* wsp = ws + (size_t)blk * SSIZE;

    // S = phi(K)^T * V : wave owns output rows e in [16w, 16w+16)
    for (int n = 0; n < 4; ++n) {
        v8f acc = {};
#pragma unroll
        for (int kk = 0; kk < CHUNK; kk += 32) {
            v16h a  = load_a_row(Kt, LDH, wave * 16, kk, lane); // (e,t) = Kt[e][t]
            v16h bb = load_b_col(Vt, LDH, kk, n * 16, lane);    // (t,d) = Vt[d][t]
            acc = WMMA_F16(acc, a, bb);
        }
        int d  = n * 16 + (lane & 15);
        int hi = lane >> 4;
#pragma unroll
        for (int r = 0; r < 8; ++r) {
            int e = wave * 16 + r + 8 * hi;
            wsp[e * Edim + d] = acc[r];
        }
    }
    // ksum[e] = sum_t phi(K)[t][e]  (contiguous row of Kt)
    for (int e = tid; e < Edim; e += 128) {
        float s = 0.f;
        for (int t = 0; t < CHUNK; ++t) s += (float)Kt[e * LDH + t];
        wsp[Edim * Edim + e] = s;
    }
}

// ---- Pass 2: in-place exclusive scan of chunk states per head -------------
__global__ __launch_bounds__(256) void la_scan(float* __restrict__ ws) {
    int bh = blockIdx.x;
    for (int i = threadIdx.x; i < SSIZE; i += 256) {
        float acc = 0.f;
        float* p = ws + (size_t)bh * NC * SSIZE + i;
        for (int c = 0; c < NC; ++c) {
            float cur = *p;
            *p = acc;          // exclusive prefix
            acc += cur;
            p += SSIZE;
        }
    }
}

// ---- Pass 3: out = [tril(QK^T) V + Q S_prev] / (rowsum + Q.kprev + eps) ---
__global__ __launch_bounds__(128) void la_output(const float* __restrict__ Qg,
                                                 const float* __restrict__ Kg,
                                                 const float* __restrict__ Vg,
                                                 const float* __restrict__ ws,
                                                 float* __restrict__ Og) {
    __shared__ alignas(16) _Float16 Ql[CHUNK * LDH];  // Ql[t][e] = phi(Q)
    __shared__ alignas(16) _Float16 Kl[CHUNK * LDH];  // Kl[t][e] = phi(K)
    __shared__ alignas(16) _Float16 Vt[Edim * LDH];   // Vt[d][t] = V[t][d]
    __shared__ alignas(16) _Float16 Al[CHUNK * LDH];  // Al[t][t'] masked scores
    __shared__ alignas(16) _Float16 St[Edim * LDH];   // St[d][e] = S_prev[e][d]
    __shared__ float kprev[Edim];
    __shared__ float invden[CHUNK];

    int blk = blockIdx.x;
    int c = blk % NC, bh = blk / NC;
    int b = bh / Hdim, h = bh % Hdim;
    int t0 = c * CHUNK;
    int tid = threadIdx.x;
    size_t base = ((size_t)(b * Tdim + t0) * Hdim + h) * Edim;
    const float* wsp = ws + (size_t)blk * SSIZE;

    for (int i = tid; i < CHUNK * Edim / 4; i += 128) {
        int t = i >> 4, e4 = (i & 15) * 4;
        size_t g = base + (size_t)t * Hdim * Edim + e4;
        const float4 qq = *(const float4*)(Qg + g);
        const float4 kq = *(const float4*)(Kg + g);
        const float4 vq = *(const float4*)(Vg + g);
        const float4 sq = *(const float4*)(wsp + (size_t)t * Edim + e4); // S_prev[t(=e)][e4(=d)]
        Ql[t * LDH + e4 + 0] = (_Float16)phi_map(qq.x);
        Ql[t * LDH + e4 + 1] = (_Float16)phi_map(qq.y);
        Ql[t * LDH + e4 + 2] = (_Float16)phi_map(qq.z);
        Ql[t * LDH + e4 + 3] = (_Float16)phi_map(qq.w);
        Kl[t * LDH + e4 + 0] = (_Float16)phi_map(kq.x);
        Kl[t * LDH + e4 + 1] = (_Float16)phi_map(kq.y);
        Kl[t * LDH + e4 + 2] = (_Float16)phi_map(kq.z);
        Kl[t * LDH + e4 + 3] = (_Float16)phi_map(kq.w);
        Vt[(e4 + 0) * LDH + t] = (_Float16)vq.x;   // transpose-scatter
        Vt[(e4 + 1) * LDH + t] = (_Float16)vq.y;
        Vt[(e4 + 2) * LDH + t] = (_Float16)vq.z;
        Vt[(e4 + 3) * LDH + t] = (_Float16)vq.w;
        St[(e4 + 0) * LDH + t] = (_Float16)sq.x;   // St[d][e]
        St[(e4 + 1) * LDH + t] = (_Float16)sq.y;
        St[(e4 + 2) * LDH + t] = (_Float16)sq.z;
        St[(e4 + 3) * LDH + t] = (_Float16)sq.w;
    }
    for (int e = tid; e < Edim; e += 128) kprev[e] = wsp[Edim * Edim + e];
    __syncthreads();

    int wave = tid >> 5, lane = tid & 31;
    int hi = lane >> 4;

    // A = tril(phi(Q) phi(K)^T); wave owns rows [16w, 16w+16)
    for (int n = 0; n < 4; ++n) {
        int col = n * 16 + (lane & 15);
        if (n > wave) { // strictly above diagonal block: zero
#pragma unroll
            for (int r = 0; r < 8; ++r)
                Al[(wave * 16 + r + 8 * hi) * LDH + col] = (_Float16)0.f;
        } else {
            v8f acc = {};
#pragma unroll
            for (int kk = 0; kk < Edim; kk += 32) {
                v16h a  = load_a_row(Ql, LDH, wave * 16, kk, lane);
                v16h bb = load_b_col(Kl, LDH, kk, n * 16, lane); // (e,t') = Kl[t'][e]
                acc = WMMA_F16(acc, a, bb);
            }
#pragma unroll
            for (int r = 0; r < 8; ++r) {
                int row = wave * 16 + r + 8 * hi;
                float v = (n < wave || col <= row) ? acc[r] : 0.f; // causal mask
                Al[row * LDH + col] = (_Float16)v;
            }
        }
    }
    __syncthreads();

    // inv denominator: rowsum(A) == phi(q_t).cumsum(K)_t (incl. diagonal)
    for (int t = tid; t < CHUNK; t += 128) {
        float s = 0.f;
        for (int j = 0; j < CHUNK; ++j) s += (float)Al[t * LDH + j];
        float dq = 0.f;
        for (int e = 0; e < Edim; ++e) dq += (float)Ql[t * LDH + e] * kprev[e];
        invden[t] = 1.f / (s + dq + EPSC);
    }
    __syncthreads();

    // out = A*V + Q*S_prev, then scale & store
    for (int n = 0; n < 4; ++n) {
        v8f acc = {};
#pragma unroll
        for (int kk = 0; kk < CHUNK; kk += 32) {   // intra-chunk: contract over t'
            v16h a  = load_a_row(Al, LDH, wave * 16, kk, lane);
            v16h bb = load_b_col(Vt, LDH, kk, n * 16, lane);  // (t',d) = Vt[d][t']
            acc = WMMA_F16(acc, a, bb);
        }
#pragma unroll
        for (int kk = 0; kk < Edim; kk += 32) {    // inter-chunk: contract over e
            v16h a  = load_a_row(Ql, LDH, wave * 16, kk, lane);
            v16h bb = load_b_col(St, LDH, kk, n * 16, lane);  // (e,d) = St[d][e]
            acc = WMMA_F16(acc, a, bb);
        }
        int d = n * 16 + (lane & 15);
#pragma unroll
        for (int r = 0; r < 8; ++r) {
            int t = wave * 16 + r + 8 * hi;
            Og[base + (size_t)t * Hdim * Edim + d] = acc[r] * invden[t];
        }
    }
}

extern "C" void kernel_launch(void* const* d_in, const int* in_sizes, int n_in,
                              void* d_out, int out_size, void* d_ws, size_t ws_size,
                              hipStream_t stream) {
    (void)in_sizes; (void)n_in; (void)out_size; (void)ws_size;
    const float* q = (const float*)d_in[0];
    const float* k = (const float*)d_in[1];
    const float* v = (const float*)d_in[2];
    float* out = (float*)d_out;
    float* ws  = (float*)d_ws;   // needs NH*NC*SSIZE*4 = 16.25 MB

    la_chunk_sums<<<dim3(NH * NC), dim3(128), 0, stream>>>(k, v, ws);
    la_scan<<<dim3(NH), dim3(256), 0, stream>>>(ws);
    la_output<<<dim3(NH * NC), dim3(128), 0, stream>>>(q, k, v, ws, out);
}
